// Net_38165079392273
// MI455X (gfx1250) — compile-verified, matrix operand order checked by reading, not run on hardware
//
#include <hip/hip_runtime.h>

// ---------------------------------------------------------------------------
// MI455X / gfx1250 implementation. Wave32, WMMA bf16 16x16x32, f32 accum.
// GEMM tiles 128x128x64, both LDS tiles stored N-major so every WMMA fragment
// is two ds_load_b128; B tile k-index XOR-swizzled to avoid store conflicts.
// ---------------------------------------------------------------------------

typedef __bf16 bf16;
typedef __attribute__((ext_vector_type(16))) __bf16 v16bf;
typedef __attribute__((ext_vector_type(8)))  float  v8f;

#define DEV __device__ __forceinline__

static constexpr int BATCH = 256;
static constexpr int DMODEL = 512;
static constexpr int NHEAD = 8;
static constexpr int NLAYER = 8;
static constexpr int TEMB = 71;   // embedding sequence length
static constexpr int TPEMB = 80;  // padded to tile multiple
static constexpr int TLOOP = 64;
static constexpr int LDT = 72;    // LDS row stride for 64-wide k tiles (bf16)

DEV bf16 f2bf(float f) {
  union { float f; unsigned u; } a; a.f = f;
  unsigned r = (a.u + 0x7FFFu + ((a.u >> 16) & 1u)) >> 16;
  union { unsigned short s; bf16 b; } o; o.s = (unsigned short)r;
  return o.b;
}
DEV float bf2f(bf16 b) {
  union { unsigned short s; bf16 b; } i; i.b = b;
  union { float f; unsigned u; } o; o.u = ((unsigned)i.s) << 16;
  return o.f;
}
DEV float lrelu(float v) { return v > 0.f ? v : 0.2f * v; }
DEV v8f zero8() { v8f z; for (int e = 0; e < 8; ++e) z[e] = 0.f; return z; }

DEV v8f wmma_bf16(v16bf a, v16bf b, v8f c) {
  return __builtin_amdgcn_wmma_f32_16x16x32_bf16(false, a, false, b, (short)0, c,
                                                 false, false);
}

// A fragment (16x32 bf16) from a row-major LDS tile (ISA 7.12.2 layout):
// lane L: row = L&15, K-half select on L>>4; two contiguous 16B blocks.
DEV v16bf fragA_rowmajor(const bf16* s, int ld, int row0, int k0) {
  int lane = threadIdx.x & 31;
  const bf16* p = s + (row0 + (lane & 15)) * ld + k0 + ((lane >> 4) << 3);
  v16bf f;
#pragma unroll
  for (int j = 0; j < 4; ++j) { f[2 * j] = p[2 * j]; f[2 * j + 1] = p[2 * j + 1]; }
#pragma unroll
  for (int j = 0; j < 4; ++j) { f[8 + 2 * j] = p[16 + 2 * j]; f[9 + 2 * j] = p[17 + 2 * j]; }
  return f;
}

// B fragment from an N-major tile with k XOR-swizzled by ((n>>4)&7)<<3.
// row0 is a multiple of 16 so all 16 lanes share one perm -> reads stay two
// aligned ds_load_b128 per lane, conflict-free.
DEV v16bf fragB_swz(const bf16* s, int ld, int row0, int k0) {
  int lane = threadIdx.x & 31;
  int row = row0 + (lane & 15);
  int kh = k0 + ((lane >> 4) << 3);
  int perm = ((row >> 4) & 7) << 3;
  const bf16* p0 = s + row * ld + (kh ^ perm);
  const bf16* p1 = s + row * ld + ((kh + 16) ^ perm);
  v16bf f;
#pragma unroll
  for (int j = 0; j < 4; ++j) { f[2 * j] = p0[2 * j]; f[2 * j + 1] = p0[2 * j + 1]; }
#pragma unroll
  for (int j = 0; j < 4; ++j) { f[8 + 2 * j] = p1[2 * j]; f[9 + 2 * j] = p1[2 * j + 1]; }
  return f;
}

// ---------------------------------------------------------------------------
__global__ __launch_bounds__(256) void cvt_bf16_kernel(const float* __restrict__ src,
                                                       bf16* __restrict__ dst, long n) {
  long i = (long)blockIdx.x * 256 + threadIdx.x;
  if (i < n) dst[i] = f2bf(src[i]);
}

// ---------------------------------------------------------------------------
// Embedding gather:  pre[b, i(71), d] f32
// ---------------------------------------------------------------------------
__global__ __launch_bounds__(256) void embed_pre_kernel(
    const int* __restrict__ fen, const int* __restrict__ move,
    const float* __restrict__ rank_emb, const float* __restrict__ file_emb,
    const float* __restrict__ fen_emb, const float* __restrict__ move_emb,
    const float* __restrict__ abs_emb, float* __restrict__ pre) {
  int bi = blockIdx.x;
  int b = bi / TEMB, i = bi - b * TEMB;
  for (int d = threadIdx.x; d < DMODEL; d += 256) {
    float v;
    if (i < 64) {
      float pos = rank_emb[(i >> 3) * DMODEL + d] + file_emb[(i & 7) * DMODEL + d];
      int f1 = fen[b * 133 + i], f2 = fen[b * 133 + 64 + i];
      v = 0.5f * (fen_emb[f1 * DMODEL + d] + fen_emb[f2 * DMODEL + d] + pos);
    } else if (i < 69) {
      int f1 = fen[b * 133 + 128 + (i - 64)];
      v = fen_emb[f1 * DMODEL + d];
    } else {
      int mi = i - 69;
      int sq = move[b * 2 + mi];
      float pos = rank_emb[(sq >> 3) * DMODEL + d] + file_emb[(sq & 7) * DMODEL + d];
      v = (pos + move_emb[mi * DMODEL + d]) * 0.58f;
    }
    v += abs_emb[i * DMODEL + d];
    pre[((long)b * TEMB + i) * DMODEL + d] = v;
  }
}

// ---------------------------------------------------------------------------
// LayerNorm over T=71 per (b,d); wave per row.
// ---------------------------------------------------------------------------
__global__ __launch_bounds__(256) void ln71_kernel(
    const float* __restrict__ pre, const float* __restrict__ lnw,
    const float* __restrict__ lnb, float* __restrict__ x71f,
    bf16* __restrict__ x71b) {
  long row = (long)blockIdx.x * 8 + (threadIdx.x >> 5);  // b*512 + d
  int lane = threadIdx.x & 31;
  long b = row >> 9;
  int d = (int)(row & 511);
  const float* p = pre + (b * TEMB) * DMODEL + d;
  float v[3];
  float s = 0.f;
#pragma unroll
  for (int q = 0; q < 3; ++q) {
    int t = lane + 32 * q;
    v[q] = (t < TEMB) ? p[(long)t * DMODEL] : 0.f;
    s += v[q];
  }
#pragma unroll
  for (int off = 16; off > 0; off >>= 1) s += __shfl_xor(s, off, 32);
  float mu = s / (float)TEMB;
  float s2 = 0.f;
#pragma unroll
  for (int q = 0; q < 3; ++q) {
    int t = lane + 32 * q;
    float dd = (t < TEMB) ? (v[q] - mu) : 0.f;
    s2 += dd * dd;
  }
#pragma unroll
  for (int off = 16; off > 0; off >>= 1) s2 += __shfl_xor(s2, off, 32);
  float inv = rsqrtf(s2 / (float)TEMB + 1e-5f);
  float* xo = x71f + row * TEMB;
  bf16* xb = x71b + row * TEMB;
#pragma unroll
  for (int q = 0; q < 3; ++q) {
    int t = lane + 32 * q;
    if (t < TEMB) {
      float nv = (v[q] - mu) * inv * lnw[t] + lnb[t];
      xo[t] = nv;
      xb[t] = f2bf(lrelu(nv));
    }
  }
}

// ---------------------------------------------------------------------------
// QKV projection GEMM: Y[r=(b,t)][c=(g,dh)] = lrelu(x)^T @ W.
// x: (B,512,Tstride) bf16 (pre-activated). W: (24,512,64) bf16.
// y: (24,B,Tp,64) bf16. 128x128x64 tiles, 8 waves (2x4), wave = 64x32.
// ---------------------------------------------------------------------------
__global__ __launch_bounds__(256) void qkv_gemm_kernel(
    const bf16* __restrict__ xb, int Tstride, int Tvalid, int Tp,
    const bf16* __restrict__ w, bf16* __restrict__ y) {
  __shared__ bf16 As[128][LDT];
  __shared__ bf16 Bs[128][LDT];  // N-major, k swizzled
  const int tid = threadIdx.x;
  const int wave = tid >> 5, lane = tid & 31;
  const int m0 = blockIdx.x * 128;
  const int n0 = blockIdx.y * 128;
  const int wm = (wave & 1) * 64;
  const int wn = (wave >> 1) * 32;
  v8f acc[4][2];
#pragma unroll
  for (int i = 0; i < 4; ++i)
#pragma unroll
    for (int j = 0; j < 2; ++j) acc[i][j] = zero8();
  bf16 zb = f2bf(0.f);

  // A slow-path map (embedding, Tstride=71): row-major per-thread run.
  const int mmA = tid >> 1;
  const int kA0 = (tid & 1) * 32;
  int rA = m0 + mmA;
  int bbA = rA / Tp, tA = rA - bbA * Tp;
  bool vA = (tA < Tvalid);
  const bf16* aSrc = xb + (long)bbA * DMODEL * Tstride + tA;
  // B map: 2 k-rows x 16 n per thread, vectorized global reads.
  const int kB0 = (tid >> 3) << 1;      // 0..62
  const int nB0 = (tid & 7) * 16;
  const int permB = ((nB0 >> 4) & 7) << 3;
  const int cB = n0 + nB0;
  const int gB = cB >> 6, dhB = cB & 63;
  const bf16* bSrc = w + ((long)gB * DMODEL) * 64 + dhB;

  for (int kt = 0; kt < DMODEL; kt += 64) {
    __syncthreads();
    if (Tstride == 64) {
      // fast path: vectorized transpose staging (Tp==Tvalid==64)
#pragma unroll
      for (int it = 0; it < 4; ++it) {
        int c = tid + it * 256;
        int kk = c & 63, rb = c >> 6;
        int r0 = m0 + rb * 8;
        int bb = r0 >> 6, t0 = r0 & 63;
        uint4 av = *(const uint4*)(xb + ((long)bb * DMODEL + kt + kk) * 64 + t0);
        const bf16* e = (const bf16*)&av;
#pragma unroll
        for (int q = 0; q < 8; ++q) As[rb * 8 + q][kk] = e[q];
      }
    } else {
      const bf16* p = aSrc + (long)(kt + kA0) * Tstride;
      bf16* dst = &As[mmA][kA0];
      if (vA) {
#pragma unroll
        for (int q = 0; q < 32; ++q) dst[q] = p[(long)q * Tstride];
      } else {
#pragma unroll
        for (int q = 0; q < 32; ++q) dst[q] = zb;
      }
    }
#pragma unroll
    for (int kq = 0; kq < 2; ++kq) {
      int d = kt + kB0 + kq;
      const bf16* src = bSrc + (long)d * 64;
      uint4 w0 = *(const uint4*)src;
      uint4 w1 = *(const uint4*)(src + 8);
      const bf16* e0 = (const bf16*)&w0;
      const bf16* e1 = (const bf16*)&w1;
      int kidx = (kB0 + kq) ^ permB;
#pragma unroll
      for (int q = 0; q < 8; ++q) Bs[nB0 + q][kidx] = e0[q];
#pragma unroll
      for (int q = 0; q < 8; ++q) Bs[nB0 + 8 + q][kidx] = e1[q];
    }
    if (kt + 64 < DMODEL) {
      __builtin_prefetch((const void*)(bSrc + (long)(kt + 64 + kB0) * 64), 0, 0);
    }
    __syncthreads();
#pragma unroll
    for (int ks = 0; ks < 64; ks += 32) {
      v16bf b0 = fragB_swz(&Bs[0][0], LDT, wn, ks);
      v16bf b1 = fragB_swz(&Bs[0][0], LDT, wn + 16, ks);
#pragma unroll
      for (int i = 0; i < 4; ++i) {
        v16bf a = fragA_rowmajor(&As[0][0], LDT, wm + i * 16, ks);
        acc[i][0] = wmma_bf16(a, b0, acc[i][0]);
        acc[i][1] = wmma_bf16(a, b1, acc[i][1]);
      }
    }
  }
#pragma unroll
  for (int i = 0; i < 4; ++i) {
    int mb = m0 + wm + i * 16 + ((lane >> 4) << 3);
    int bb = mb / Tp, t0 = mb - bb * Tp;  // 8-run stays in one b (Tp%16==0)
#pragma unroll
    for (int j = 0; j < 2; ++j) {
      int n = n0 + wn + j * 16 + (lane & 15);
      int g = n >> 6, dh = n & 63;
      bf16* yp = y + (((long)g * BATCH + bb) * Tp + t0) * 64 + dh;
#pragma unroll
      for (int r = 0; r < 8; ++r) yp[r * 64] = f2bf(acc[i][j][r]);
    }
  }
}

// ---------------------------------------------------------------------------
// RMS over token axis per (g,b,dh). y,qn: (24,B,Tp,64) bf16.
// ---------------------------------------------------------------------------
__global__ __launch_bounds__(256) void rms_tok_kernel(
    const bf16* __restrict__ y, bf16* __restrict__ qn, int Tp, int Tvalid,
    const float* __restrict__ wptr, int widx) {
  int grp = blockIdx.x * 4 + (threadIdx.x >> 6);  // over 24*B
  int dh = threadIdx.x & 63;
  const bf16* p = y + ((long)grp * Tp) * 64 + dh;
  float s = 0.f;
  for (int t = 0; t < Tvalid; ++t) {
    float v = bf2f(p[t * 64]);
    s += v * v;
  }
  float r = wptr[widx] * rsqrtf(s / (float)Tvalid + 1e-6f);
  bf16* q = qn + ((long)grp * Tp) * 64 + dh;
  for (int t = 0; t < Tvalid; ++t) q[t * 64] = f2bf(bf2f(p[t * 64]) * r);
}

// ---------------------------------------------------------------------------
// Attention per (b,h), LDS-resident. S=QK^T/8 (WMMA), masked softmax,
// O=P@V (WMMA, V transposed once into v_t so both GEMMs use b128 fragments).
// Epilogue: residual add -> xf f32, xb = lrelu(xf) bf16.
// ---------------------------------------------------------------------------
__global__ __launch_bounds__(256) void attention_kernel(
    const bf16* __restrict__ qn, int Tp, int Tvalid, const float* resid,
    int residT, float* xf, bf16* xbo) {
  __shared__ __align__(16) char smem[63232];
  bf16* q_s = (bf16*)smem;              // [80][72]
  bf16* k_s = (bf16*)(smem + 11520);    // [80][72]
  bf16* v_t = (bf16*)(smem + 23040);    // [64][104]  (rows = dh, cols = t')
  float* S_s = (float*)(smem + 36352);  // [80][84]
  bf16* P_s = (bf16*)smem;              // [80][104]  aliases q_s/k_s after use

  const int tid = threadIdx.x;
  const int wave = tid >> 5, lane = tid & 31;
  const int b = blockIdx.x >> 3;
  const int h = blockIdx.x & 7;
  const bf16* qp = qn + ((long)(h)*BATCH + b) * Tp * 64;
  const bf16* kp = qn + ((long)(8 + h) * BATCH + b) * Tp * 64;
  const bf16* vp = qn + ((long)(16 + h) * BATCH + b) * Tp * 64;
  bf16 zb = f2bf(0.f);
  uint4 z4 = make_uint4(0u, 0u, 0u, 0u);
  for (int c = tid; c < 80 * 8; c += 256) {
    int t = c >> 3, col = (c & 7) << 3;
    uint4 qv = z4, kv = z4;
    if (t < Tvalid) {
      qv = *(const uint4*)(qp + t * 64 + col);
      kv = *(const uint4*)(kp + t * 64 + col);
    }
    *(uint4*)(q_s + t * 72 + col) = qv;
    *(uint4*)(k_s + t * 72 + col) = kv;
  }
  for (int c = tid; c < 96 * 8; c += 256) {
    int t = c >> 3, dhb = (c & 7) << 3;
    uint4 vv = z4;
    if (t < Tvalid) vv = *(const uint4*)(vp + t * 64 + dhb);
    const bf16* e = (const bf16*)&vv;
#pragma unroll
    for (int q = 0; q < 8; ++q) v_t[(dhb + q) * 104 + t] = e[q];
  }
  __syncthreads();

  // S = Q K^T * 0.125 ; B^T fragment of row-major K == A-style fragment.
  for (int tile = wave; tile < 25; tile += 8) {
    int i = tile / 5, j = tile - i * 5;
    v8f acc = zero8();
#pragma unroll
    for (int ks = 0; ks < 64; ks += 32) {
      v16bf a = fragA_rowmajor(q_s, 72, i * 16, ks);
      v16bf bb = fragA_rowmajor(k_s, 72, j * 16, ks);
      acc = wmma_bf16(a, bb, acc);
    }
    int n = j * 16 + (lane & 15);
    int rbase = i * 16 + ((lane >> 4) << 3);
#pragma unroll
    for (int r = 0; r < 8; ++r) S_s[(rbase + r) * 84 + n] = acc[r] * 0.125f;
  }
  __syncthreads();

  for (int row = wave; row < 80; row += 8) {
    if (row >= Tvalid) {
      for (int c = lane; c < 96; c += 32) P_s[row * 104 + c] = zb;
    } else {
      float v0[3];
      float mx = -1e30f;
#pragma unroll
      for (int q = 0; q < 3; ++q) {
        int c = lane + 32 * q;
        v0[q] = (c < Tvalid) ? S_s[row * 84 + c] : -1e30f;
        mx = fmaxf(mx, v0[q]);
      }
#pragma unroll
      for (int off = 16; off > 0; off >>= 1) mx = fmaxf(mx, __shfl_xor(mx, off, 32));
      float sum = 0.f;
      float e0[3];
#pragma unroll
      for (int q = 0; q < 3; ++q) {
        int c = lane + 32 * q;
        e0[q] = (c < Tvalid) ? __expf(v0[q] - mx) : 0.f;
        sum += e0[q];
      }
#pragma unroll
      for (int off = 16; off > 0; off >>= 1) sum += __shfl_xor(sum, off, 32);
      float inv = 1.f / sum;
#pragma unroll
      for (int q = 0; q < 3; ++q) {
        int c = lane + 32 * q;
        P_s[row * 104 + c] = f2bf(e0[q] * inv);
      }
    }
  }
  __syncthreads();

  // O = P @ V : A from P_s (row-major), B from v_t (N-major rows = dh).
  for (int tile = wave; tile < 20; tile += 8) {
    int i = tile >> 2, j = tile & 3;
    v8f acc = zero8();
#pragma unroll
    for (int ks = 0; ks < 96; ks += 32) {
      v16bf a = fragA_rowmajor(P_s, 104, i * 16, ks);
      v16bf bb = fragA_rowmajor(v_t, 104, j * 16, ks);
      acc = wmma_bf16(a, bb, acc);
    }
    int dh = j * 16 + (lane & 15);
    int d = h * 64 + dh;
    int tb = i * 16 + ((lane >> 4) << 3);
#pragma unroll
    for (int r = 0; r < 8; ++r) {
      int t = tb + r;
      if (t < 64) {
        long rowa = (long)b * DMODEL + d;
        float nv = resid[rowa * residT + t] + acc[r];
        long oa = rowa * 64 + t;
        xf[oa] = nv;
        xbo[oa] = f2bf(lrelu(nv));
      }
    }
  }
}

// ---------------------------------------------------------------------------
// 3x3 SAME conv as implicit GEMM, 128x128x64 tiles.
// A = W[OC][IC*9] (contiguous k), B = on-the-fly im2col of xin (B,IC,8,8).
// mode 0: yout = acc+bias.  mode 1: xf += acc+bias; xb = lrelu(xf).
// ---------------------------------------------------------------------------
__global__ __launch_bounds__(256) void conv_gemm_kernel(
    const bf16* __restrict__ xin, int IC, const bf16* __restrict__ w,
    const float* __restrict__ bias, int OC, float* yout, const float* resid,
    bf16* xbo, int mode) {
  __shared__ bf16 As[128][LDT];
  __shared__ bf16 Bs[128][LDT];  // N-major, k swizzled
  const int tid = threadIdx.x;
  const int wave = tid >> 5, lane = tid & 31;
  const int m0 = blockIdx.x * 128;  // oc
  const int n0 = blockIdx.y * 128;  // b*64 + s
  const int wm = (wave & 1) * 64;
  const int wn = (wave >> 1) * 32;
  const int K = IC * 9;
  v8f acc[4][2];
#pragma unroll
  for (int i = 0; i < 4; ++i)
#pragma unroll
    for (int j = 0; j < 2; ++j) acc[i][j] = zero8();
  bf16 zb = f2bf(0.f);
  const int mmA = tid >> 1;
  const int kA0 = (tid & 1) * 32;
  const bf16* aSrc = w + (long)(m0 + mmA) * K + kA0;
  const int kB0 = (tid >> 3) << 1;
  const int nB0 = (tid & 7) * 16;
  const int permB = ((nB0 >> 4) & 7) << 3;
  const int nAbs = n0 + nB0;          // multiple of 16 -> same b, ix0 = 0
  const int bbB = nAbs >> 6;
  const int iy0 = (nAbs & 63) >> 3;
  const long xBase = (long)bbB * IC * 64;

  for (int kt = 0; kt < K; kt += 64) {
    __syncthreads();
    {
      const bf16* wp = aSrc + kt;  // 32 contiguous bf16, 16B aligned
      bf16* dst = &As[mmA][kA0];
#pragma unroll
      for (int q = 0; q < 32; ++q) dst[q] = wp[q];
    }
#pragma unroll
    for (int kq = 0; kq < 2; ++kq) {
      int k = kt + kB0 + kq;
      int ic = k / 9;
      int p = k - ic * 9;
      int ky = p / 3;
      int kx = p - ky * 3;
      const bf16* src = xin + xBase + (long)ic * 64;
      int kidx = (kB0 + kq) ^ permB;
#pragma unroll
      for (int q = 0; q < 16; ++q) {
        int iy = iy0 + (q >> 3) + ky - 1;
        int ix = (q & 7) + kx - 1;
        bf16 v = zb;
        if ((unsigned)iy < 8u && (unsigned)ix < 8u) v = src[iy * 8 + ix];
        Bs[nB0 + q][kidx] = v;
      }
    }
    if (kt + 64 < K) {
      __builtin_prefetch((const void*)(aSrc + kt + 64), 0, 0);
    }
    __syncthreads();
#pragma unroll
    for (int ks = 0; ks < 64; ks += 32) {
      v16bf b0 = fragB_swz(&Bs[0][0], LDT, wn, ks);
      v16bf b1 = fragB_swz(&Bs[0][0], LDT, wn + 16, ks);
#pragma unroll
      for (int i = 0; i < 4; ++i) {
        v16bf a = fragA_rowmajor(&As[0][0], LDT, wm + i * 16, ks);
        acc[i][0] = wmma_bf16(a, b0, acc[i][0]);
        acc[i][1] = wmma_bf16(a, b1, acc[i][1]);
      }
    }
  }
#pragma unroll
  for (int i = 0; i < 4; ++i) {
    int mb = m0 + wm + i * 16 + ((lane >> 4) << 3);
#pragma unroll
    for (int j = 0; j < 2; ++j) {
      int n = n0 + wn + j * 16 + (lane & 15);
      int bb = n >> 6, s = n & 63;
#pragma unroll
      for (int r = 0; r < 8; ++r) {
        int oc = mb + r;
        float v = acc[i][j][r] + bias[oc];
        long addr = ((long)bb * OC + oc) * 64 + s;
        if (mode == 0) {
          yout[addr] = v;
        } else {
          float nv = resid[addr] + v;
          yout[addr] = nv;
          xbo[addr] = f2bf(lrelu(nv));
        }
      }
    }
  }
}

// ---------------------------------------------------------------------------
// Spatial RMS over the 64 positions per (b,c), then lrelu, -> bf16.
// ---------------------------------------------------------------------------
__global__ __launch_bounds__(256) void rms_spatial_kernel(
    const float* __restrict__ yin, const float* __restrict__ nw,
    bf16* __restrict__ xbout) {
  long row = (long)blockIdx.x * 8 + (threadIdx.x >> 5);  // b*C + c
  int lane = threadIdx.x & 31;
  const float* p = yin + row * 64;
  float v0 = p[lane], v1 = p[lane + 32];
  float s = v0 * v0 + v1 * v1;
#pragma unroll
  for (int off = 16; off > 0; off >>= 1) s += __shfl_xor(s, off, 32);
  float r = nw[0] * rsqrtf(s / 64.f + 1e-6f);
  bf16* q = xbout + row * 64;
  q[lane] = f2bf(lrelu(v0 * r));
  q[lane + 32] = f2bf(lrelu(v1 * r));
}

// ---------------------------------------------------------------------------
// Final 8x8 VALID conv: out[b] = dot(ub[b], w2) + b2.
// ---------------------------------------------------------------------------
__global__ __launch_bounds__(256) void final_dot_kernel(
    const bf16* __restrict__ ub, const bf16* __restrict__ w2,
    const float* __restrict__ b2, float* __restrict__ out) {
  int b = blockIdx.x;
  const bf16* p = ub + (long)b * 65536;
  float s = 0.f;
  for (int i = threadIdx.x; i < 65536; i += 256) s += bf2f(p[i]) * bf2f(w2[i]);
#pragma unroll
  for (int off = 16; off > 0; off >>= 1) s += __shfl_xor(s, off, 32);
  __shared__ float red[8];
  if ((threadIdx.x & 31) == 0) red[threadIdx.x >> 5] = s;
  __syncthreads();
  if (threadIdx.x == 0) {
    float t = 0.f;
    for (int k = 0; k < 8; ++k) t += red[k];
    out[b] = t + b2[0];
  }
}

// ---------------------------------------------------------------------------
// Host driver
// ---------------------------------------------------------------------------
extern "C" void kernel_launch(void* const* d_in, const int* in_sizes, int n_in,
                              void* d_out, int out_size, void* d_ws, size_t ws_size,
                              hipStream_t stream) {
  const int* fen = (const int*)d_in[0];
  const int* move = (const int*)d_in[1];
  const float* rank_emb = (const float*)d_in[2];
  const float* file_emb = (const float*)d_in[3];
  const float* fen_emb = (const float*)d_in[4];
  const float* move_emb = (const float*)d_in[5];
  const float* abs_emb = (const float*)d_in[6];
  const float* ln_w = (const float*)d_in[7];
  const float* ln_b = (const float*)d_in[8];
  const float* emb_qkv = (const float*)d_in[9];
  const float* emb_norm_w = (const float*)d_in[10];
  const float* qkv = (const float*)d_in[11];
  const float* mha_norm_w = (const float*)d_in[12];
  const float* conv1_w = (const float*)d_in[13];
  const float* conv1_b = (const float*)d_in[14];
  const float* conv2_w = (const float*)d_in[15];
  const float* conv2_b = (const float*)d_in[16];
  const float* cb_norm_w = (const float*)d_in[17];
  const float* out_norm_w = (const float*)d_in[18];
  const float* co1_w = (const float*)d_in[19];
  const float* co1_b = (const float*)d_in[20];
  const float* co2_w = (const float*)d_in[21];
  const float* co2_b = (const float*)d_in[22];
  float* out = (float*)d_out;

  char* ws = (char*)d_ws;
  size_t off = 0;
  auto carve = [&](size_t bytes) -> char* {
    char* p = ws + off;
    off = (off + bytes + 255) & ~(size_t)255;
    return p;
  };

  const long NC1 = (long)NLAYER * 512 * 512 * 9;
  const long NQE = 3L * 8 * 512 * 64;
  const long NQL = (long)NLAYER * NQE;
  const long NO1 = 1024L * 512 * 9;
  const long NO2 = 1024L * 64;

  bf16* wc1 = (bf16*)carve(NC1 * 2);
  bf16* wc2 = (bf16*)carve(NC1 * 2);
  bf16* wqe = (bf16*)carve(NQE * 2);
  bf16* wql = (bf16*)carve(NQL * 2);
  bf16* wo1 = (bf16*)carve(NO1 * 2);
  bf16* wo2 = (bf16*)carve(NO2 * 2);
  char* slot_pre = carve((size_t)BATCH * TEMB * DMODEL * 4);  // pre | t1f
  float* pre = (float*)slot_pre;
  float* t1f = (float*)slot_pre;
  char* slot_u = carve((size_t)BATCH * 1024 * 64 * 4);  // x71f | uf
  float* x71f = (float*)slot_u;
  float* uf = (float*)slot_u;
  char* slot_ub = carve((size_t)BATCH * 1024 * 64 * 2);  // x71b | ub
  bf16* x71b = (bf16*)slot_ub;
  bf16* ub = (bf16*)slot_ub;
  float* xf = (float*)carve((size_t)BATCH * DMODEL * 64 * 4);
  bf16* xb = (bf16*)carve((size_t)BATCH * DMODEL * 64 * 2);
  bf16* t1b = (bf16*)carve((size_t)BATCH * DMODEL * 64 * 2);
  bf16* ybuf = (bf16*)carve((size_t)24 * BATCH * TPEMB * 64 * 2);
  bf16* qn = (bf16*)carve((size_t)24 * BATCH * TPEMB * 64 * 2);
  (void)ws_size;

  auto cvt = [&](const float* s, bf16* d, long n) {
    cvt_bf16_kernel<<<dim3((unsigned)((n + 255) / 256)), dim3(256), 0, stream>>>(s, d, n);
  };
  cvt(conv1_w, wc1, NC1);
  cvt(conv2_w, wc2, NC1);
  cvt(emb_qkv, wqe, NQE);
  cvt(qkv, wql, NQL);
  cvt(co1_w, wo1, NO1);
  cvt(co2_w, wo2, NO2);

  embed_pre_kernel<<<dim3(BATCH * TEMB), dim3(256), 0, stream>>>(
      fen, move, rank_emb, file_emb, fen_emb, move_emb, abs_emb, pre);
  ln71_kernel<<<dim3(BATCH * DMODEL / 8), dim3(256), 0, stream>>>(pre, ln_w, ln_b,
                                                                  x71f, x71b);

  // Embedding MHA (T=71 padded to 80); keep first 64 tokens -> xf/xb
  qkv_gemm_kernel<<<dim3(BATCH * TPEMB / 128, 12), dim3(256), 0, stream>>>(
      x71b, TEMB, TEMB, TPEMB, wqe, ybuf);
  rms_tok_kernel<<<dim3(24 * BATCH / 4), dim3(256), 0, stream>>>(
      ybuf, qn, TPEMB, TEMB, emb_norm_w, 0);
  attention_kernel<<<dim3(BATCH * NHEAD), dim3(256), 0, stream>>>(
      qn, TPEMB, TEMB, x71f, TEMB, xf, xb);

  for (int l = 0; l < NLAYER; ++l) {
    const long wstep = 512L * 512 * 9;
    conv_gemm_kernel<<<dim3(4, 128), dim3(256), 0, stream>>>(
        xb, 512, wc1 + (long)l * wstep, conv1_b + l * 512, 512, t1f, nullptr,
        nullptr, 0);
    rms_spatial_kernel<<<dim3(BATCH * DMODEL / 8), dim3(256), 0, stream>>>(
        t1f, cb_norm_w + l, t1b);
    conv_gemm_kernel<<<dim3(4, 128), dim3(256), 0, stream>>>(
        t1b, 512, wc2 + (long)l * wstep, conv2_b + l * 512, 512, xf, xf, xb, 1);
    qkv_gemm_kernel<<<dim3(BATCH * TLOOP / 128, 12), dim3(256), 0, stream>>>(
        xb, TLOOP, TLOOP, TLOOP, wql + (long)l * NQE, ybuf);
    rms_tok_kernel<<<dim3(24 * BATCH / 4), dim3(256), 0, stream>>>(
        ybuf, qn, TLOOP, TLOOP, mha_norm_w, l);
    attention_kernel<<<dim3(BATCH * NHEAD), dim3(256), 0, stream>>>(
        qn, TLOOP, TLOOP, xf, TLOOP, xf, xb);
  }

  conv_gemm_kernel<<<dim3(8, 128), dim3(256), 0, stream>>>(
      xb, 512, wo1, co1_b, 1024, uf, nullptr, nullptr, 0);
  rms_spatial_kernel<<<dim3(BATCH * 1024 / 8), dim3(256), 0, stream>>>(
      uf, out_norm_w, ub);
  final_dot_kernel<<<dim3(BATCH), dim3(256), 0, stream>>>(ub, wo2, co2_b, out);
  (void)in_sizes; (void)n_in; (void)out_size;
}
